// MLSDTA_53927609368846
// MI455X (gfx1250) — compile-verified, matrix operand order checked by reading, not run on hardware
//
#include <hip/hip_runtime.h>
#include <math.h>

// ---------------------------------------------------------------------------
// Types / helpers
// ---------------------------------------------------------------------------
typedef __attribute__((ext_vector_type(16))) _Float16 v16h;
typedef __attribute__((ext_vector_type(8)))  _Float16 v8h;
typedef __attribute__((ext_vector_type(8)))  float    v8f;

#define DEV __device__ __forceinline__

DEV unsigned fkey(float f) {            // order-preserving float->uint map
  unsigned u = __float_as_uint(f);
  return (u & 0x80000000u) ? ~u : (u | 0x80000000u);
}
DEV float funkey(unsigned k) {
  return __uint_as_float((k & 0x80000000u) ? (k & 0x7fffffffu) : ~k);
}
DEV float sigmoidf(float x) { return 1.0f / (1.0f + expf(-x)); }

static inline int cdiv(long long a, long long b) { return (int)((a + b - 1) / b); }

// ---------------------------------------------------------------------------
// WMMA GEMM staging: write one 64x32 A tile and one 32x64 B tile (transposed
// in LDS: Bs[col][k]) as f16. TAIL=false => no K guard (full tile): loads are
// unconditional (index-clamped only; M/N overrun duplicates last row/col and
// is masked at the epilogue). TAIL=true => zero K padding via x*mask so the
// compiler cannot sink loads into divergent branches.
// ---------------------------------------------------------------------------
template <int MODE, bool TAIL>
DEV void k_gemm_stage(const float* __restrict__ A, const float* __restrict__ Wt,
                      _Float16* As, _Float16* Bs, int k0, int M, int N, int K,
                      int Cin, int L, int ar_, int ak_, int bn_, int bk_,
                      int gmA, int gnB) {
  // ---- A tile ----
  {
    const int gmc = (gmA < M) ? gmA : (M - 1);
    v8h av;
#pragma unroll
    for (int i = 0; i < 8; ++i) {
      const int gk = k0 + ak_ + i;
      float v;
      if (MODE == 0) {
        const int gkc = TAIL ? ((gk < K) ? gk : (K - 1)) : gk;
        v = A[(size_t)gmc * K + gkc];
        if (TAIL) v *= (gk < K) ? 1.0f : 0.0f;
      } else {
        const int b   = gmc / L, l = gmc - b * L;
        const int gkc = TAIL ? ((gk < K) ? gk : (K - 1)) : gk;
        const int ci  = gkc / 3, t = gkc - ci * 3;
        const int pos = l + t - 1;
        const int posc = (pos < 0) ? 0 : ((pos >= L) ? (L - 1) : pos);
        v = (MODE == 1) ? A[((size_t)b * Cin + ci) * L + posc]
                        : A[((size_t)b * L + posc) * Cin + ci];
        float mask = ((unsigned)pos < (unsigned)L) ? 1.0f : 0.0f;
        if (TAIL) mask = (gk < K) ? mask : 0.0f;
        v *= mask;
      }
      av[i] = (_Float16)v;
    }
    *(v8h*)&As[ar_ * 40 + ak_] = av;
  }
  // ---- B tile (transposed into LDS) ----
  {
    const int gnc = (gnB < N) ? gnB : (N - 1);
    v8h bv;
#pragma unroll
    for (int i = 0; i < 8; ++i) {
      const int gk  = k0 + bk_ + i;
      const int gkc = TAIL ? ((gk < K) ? gk : (K - 1)) : gk;
      float v = Wt[(size_t)gkc * N + gnc];
      if (TAIL) v *= (gk < K) ? 1.0f : 0.0f;
      bv[i] = (_Float16)v;
    }
    *(v8h*)&Bs[bn_ * 40 + bk_] = bv;
  }
}

// ---------------------------------------------------------------------------
// WMMA GEMM: C[M,N] = act(A[M,K] @ W[K,N] + bias)
// MODE 0: A is plain row-major [M,K]
// MODE 1: implicit im2col, X is [B, Cin, L], K = Cin*3 (conv1d k=3, pad 1)
// MODE 2: implicit im2col, X is [B*L, Cin] (L-major rows), K = Cin*3
// Block tile 64x64, 8 wave32 waves in 4x2 grid, each wave -> 16x32 via 2 WMMAs
// ---------------------------------------------------------------------------
template <int MODE>
__global__ __launch_bounds__(256)
void k_gemm(const float* __restrict__ A, const float* __restrict__ Wt,
            const float* __restrict__ bias, float* __restrict__ C,
            int M, int N, int K, int act, int Cin, int L) {
  __shared__ __align__(16) _Float16 As[64 * 40];
  __shared__ __align__(16) _Float16 Bs[64 * 40];

  const int tid  = threadIdx.x;
  const int lane = tid & 31;
  const int wid  = tid >> 5;
  const int wm   = wid >> 1;           // 0..3
  const int wn   = wid & 1;            // 0..1
  const int bm   = blockIdx.y * 64;
  const int bn   = blockIdx.x * 64;
  const int hl   = lane >> 4;          // half of wave
  const int l16  = lane & 15;

  // staging coordinates
  const int ar_ = tid >> 2;            // A tile row 0..63
  const int ak_ = (tid & 3) * 8;       // A k-subgroup {0,8,16,24}
  const int bn_ = tid & 63;            // B tile col 0..63
  const int bk_ = (tid >> 6) * 8;      // B k-subgroup {0,8,16,24}
  const int gmA = bm + ar_;
  const int gnB = bn + bn_;

  v8f acc0 = {};
  v8f acc1 = {};

  auto do_mma = [&]() {
    __syncthreads();
    const int arow = (wm * 16 + l16) * 40;
    v8h a0 = *(const v8h*)&As[arow + hl * 8];
    v8h a1 = *(const v8h*)&As[arow + 16 + hl * 8];
    v16h af = __builtin_shufflevector(a0, a1, 0, 1, 2, 3, 4, 5, 6, 7,
                                      8, 9, 10, 11, 12, 13, 14, 15);
    const int bcol0 = (wn * 32 + l16) * 40;
    v8h b00 = *(const v8h*)&Bs[bcol0 + hl * 8];
    v8h b01 = *(const v8h*)&Bs[bcol0 + 16 + hl * 8];
    v16h bf0 = __builtin_shufflevector(b00, b01, 0, 1, 2, 3, 4, 5, 6, 7,
                                       8, 9, 10, 11, 12, 13, 14, 15);
    const int bcol1 = bcol0 + 16 * 40;
    v8h b10 = *(const v8h*)&Bs[bcol1 + hl * 8];
    v8h b11 = *(const v8h*)&Bs[bcol1 + 16 + hl * 8];
    v16h bf1 = __builtin_shufflevector(b10, b11, 0, 1, 2, 3, 4, 5, 6, 7,
                                       8, 9, 10, 11, 12, 13, 14, 15);
    acc0 = __builtin_amdgcn_wmma_f32_16x16x32_f16(false, af, false, bf0,
                                                  (short)0, acc0, false, false);
    acc1 = __builtin_amdgcn_wmma_f32_16x16x32_f16(false, af, false, bf1,
                                                  (short)0, acc1, false, false);
    __syncthreads();
  };

  const int kfull = K & ~31;           // full 32-wide tiles
  for (int k0 = 0; k0 < kfull; k0 += 32) {
    k_gemm_stage<MODE, false>(A, Wt, As, Bs, k0, M, N, K, Cin, L,
                              ar_, ak_, bn_, bk_, gmA, gnB);
    // prefetch next K tile (lowers to global_prefetch on gfx1250)
    {
      const int nk = k0 + 32;
      if (nk < K) {
        if (MODE == 0) {
          int ar = (gmA < M) ? gmA : (M - 1);
          __builtin_prefetch(&A[(size_t)ar * K + nk], 0, 1);
        }
        int br = nk + bk_; if (br >= K) br = K - 1;
        int bc = (gnB < N) ? gnB : (N - 1);
        __builtin_prefetch(&Wt[(size_t)br * N + bc], 0, 1);
      }
    }
    do_mma();
  }
  if (kfull < K) {
    k_gemm_stage<MODE, true>(A, Wt, As, Bs, kfull, M, N, K, Cin, L,
                             ar_, ak_, bn_, bk_, gmA, gnB);
    do_mma();
  }

  // ---- epilogue: C layout lane=(hl,n), VGPR v -> row hl*8+v ----
  const int m0 = bm + wm * 16 + hl * 8;
  const int n0 = bn + wn * 32 + l16;
#pragma unroll
  for (int v = 0; v < 8; ++v) {
    const int gm = m0 + v;
    if (gm < M) {
      if (n0 < N) {
        float r0 = acc0[v] + (bias ? bias[n0] : 0.0f);
        if (act == 1) r0 = fmaxf(r0, 0.0f);
        C[(size_t)gm * N + n0] = r0;
      }
      const int n1 = n0 + 16;
      if (n1 < N) {
        float r1 = acc1[v] + (bias ? bias[n1] : 0.0f);
        if (act == 1) r1 = fmaxf(r1, 0.0f);
        C[(size_t)gm * N + n1] = r1;
      }
    }
  }
}

// ---------------------------------------------------------------------------
// Elementwise / scatter / reduction kernels
// ---------------------------------------------------------------------------
__global__ void k_fill(float* p, long long n, float v) {
  long long i = (long long)blockIdx.x * blockDim.x + threadIdx.x;
  if (i < n) p[i] = v;
}
__global__ void k_init_u(unsigned* p, long long n, unsigned v) {
  long long i = (long long)blockIdx.x * blockDim.x + threadIdx.x;
  if (i < n) p[i] = v;
}
__global__ void k_decode_u(unsigned* p, long long n) {
  long long i = (long long)blockIdx.x * blockDim.x + threadIdx.x;
  if (i < n) p[i] = __float_as_uint(funkey(p[i]));
}
__global__ void k_degree(const int* col, int E, float* deg) {
  int e = blockIdx.x * blockDim.x + threadIdx.x;
  if (e < E) atomicAdd(&deg[col[e]], 1.0f);
}
__global__ void k_invsqrt(const float* deg, float* dis, int n) {
  int i = blockIdx.x * blockDim.x + threadIdx.x;
  if (i < n) dis[i] = (deg[i] > 0.0f) ? rsqrtf(deg[i]) : 0.0f;
}
__global__ void k_gcn_edges(const float* xw, const int* row, const int* col,
                            const float* dis, float* out, int E, int F) {
  long long i = (long long)blockIdx.x * blockDim.x + threadIdx.x;
  if (i >= (long long)E * F) return;
  int e = (int)(i / F), f = (int)(i % F);
  int r = row[e], c = col[e];
  atomicAdd(&out[(size_t)c * F + f], xw[(size_t)r * F + f] * dis[r] * dis[c]);
}
__global__ void k_gcn_self(const float* xw, const float* dis, const float* b,
                           float* out, int n, int F) {
  long long i = (long long)blockIdx.x * blockDim.x + threadIdx.x;
  if (i >= (long long)n * F) return;
  int node = (int)(i / F), f = (int)(i % F);
  out[i] += xw[i] * dis[node] * dis[node] + b[f];
}
__global__ void k_colmean(const float* x, float* mean, int n, int F) {
  __shared__ float red[256];
  int c = blockIdx.x;
  float s = 0.0f;
  for (int r = threadIdx.x; r < n; r += 256) s += x[(size_t)r * F + c];
  red[threadIdx.x] = s;
  __syncthreads();
  for (int st = 128; st > 0; st >>= 1) {
    if (threadIdx.x < st) red[threadIdx.x] += red[threadIdx.x + st];
    __syncthreads();
  }
  if (threadIdx.x == 0) mean[c] = red[0] / (float)n;
}
__global__ void k_pairnorm_relu(const float* x, float* y, const float* mean,
                                int n, int F, int do_relu) {
  int r = blockIdx.x * blockDim.x + threadIdx.x;
  if (r >= n) return;
  float ss = 0.0f;
  for (int f = 0; f < F; ++f) {
    float d = x[(size_t)r * F + f] - mean[f];
    ss += d * d;
  }
  float inv = 1.0f / (1e-6f + sqrtf(ss));
  for (int f = 0; f < F; ++f) {
    float v = (x[(size_t)r * F + f] - mean[f]) * inv;
    if (do_relu) v = fmaxf(v, 0.0f);
    y[(size_t)r * F + f] = v;
  }
}
__global__ void k_concat3(const float* a, const float* b, const float* c,
                          float* y, int n, int Fg) {
  long long i = (long long)blockIdx.x * blockDim.x + threadIdx.x;
  int F3 = 3 * Fg;
  if (i >= (long long)n * F3) return;
  int node = (int)(i / F3), f = (int)(i % F3);
  float v;
  if (f < Fg)            v = a[(size_t)node * Fg + f];
  else if (f < 2 * Fg)   v = b[(size_t)node * Fg + (f - Fg)];
  else                   v = c[(size_t)node * Fg + (f - 2 * Fg)];
  y[i] = v;
}
__global__ void k_segmax_feat(const float* xp, const int* row, const int* col,
                              unsigned* xq, int E, int n, int F) {
  long long i = (long long)blockIdx.x * blockDim.x + threadIdx.x;
  if (i >= (long long)(E + n) * F) return;
  int e = (int)(i / F), f = (int)(i % F);
  int r, c;
  if (e < E) { r = row[e]; c = col[e]; } else { r = c = e - E; }
  atomicMax(&xq[(size_t)c * F + f], fkey(xp[(size_t)r * F + f]));
}
__global__ void k_asap_att(const float* xq2, const float* xpool,
                           const int* row, const int* col,
                           const float* attW, const float* attb,
                           float* s, int E, int n, int F) {
  int e = blockIdx.x * blockDim.x + threadIdx.x;
  if (e >= E + n) return;
  int r, c;
  if (e < E) { r = row[e]; c = col[e]; } else { r = c = e - E; }
  float acc = attb[0];
  for (int f = 0; f < F; ++f)
    acc += xq2[(size_t)c * F + f] * attW[f] + xpool[(size_t)r * F + f] * attW[F + f];
  s[e] = (acc > 0.0f) ? acc : 0.2f * acc;   // leaky_relu 0.2
}
__global__ void k_segmax_scalar(const float* s, const int* col, unsigned* smax,
                                int E, int n) {
  int e = blockIdx.x * blockDim.x + threadIdx.x;
  if (e >= E + n) return;
  int c = (e < E) ? col[e] : e - E;
  atomicMax(&smax[c], fkey(s[e]));
}
__global__ void k_segexp(float* s, const float* smax, float* dsum,
                         const int* col, int E, int n) {
  int e = blockIdx.x * blockDim.x + threadIdx.x;
  if (e >= E + n) return;
  int c = (e < E) ? col[e] : e - E;
  float v = expf(s[e] - smax[c]);
  s[e] = v;
  atomicAdd(&dsum[c], v);
}
__global__ void k_segdiv(float* s, const float* dsum, const int* col, int E, int n) {
  int e = blockIdx.x * blockDim.x + threadIdx.x;
  if (e >= E + n) return;
  int c = (e < E) ? col[e] : e - E;
  s[e] /= dsum[c];
}
__global__ void k_weighted_scatter(const float* h, const float* s,
                                   const int* row, const int* col,
                                   float* xn, int E, int n, int F) {
  long long i = (long long)blockIdx.x * blockDim.x + threadIdx.x;
  if (i >= (long long)(E + n) * F) return;
  int e = (int)(i / F), f = (int)(i % F);
  int r, c;
  if (e < E) { r = row[e]; c = col[e]; } else { r = c = e - E; }
  atomicAdd(&xn[(size_t)c * F + f], h[(size_t)r * F + f] * s[e]);
}
__global__ void k_le_ab(const float* xn, const float* w1, const float* b1,
                        const float* w2, float* a, float* b2, int n, int F) {
  int i = blockIdx.x * blockDim.x + threadIdx.x;
  if (i >= n) return;
  float s1 = b1[0], s2 = 0.0f;
  for (int f = 0; f < F; ++f) {
    float v = xn[(size_t)i * F + f];
    s1 += v * w1[f];
    s2 += v * w2[f];
  }
  a[i] = s1;
  b2[i] = s2;
}
__global__ void k_le_agg(const float* a, const float* b2, const int* row,
                         const int* col, float* agg, int E, int n) {
  int e = blockIdx.x * blockDim.x + threadIdx.x;
  if (e >= E + n) return;
  int r, c;
  if (e < E) { r = row[e]; c = col[e]; } else { r = c = e - E; }
  atomicAdd(&agg[c], a[r] - b2[c]);
}
__global__ void k_fit(const float* xn, const float* agg, const float* w3,
                      const float* b3, float* fit, int n, int F) {
  int i = blockIdx.x * blockDim.x + threadIdx.x;
  if (i >= n) return;
  float s = agg[i] + b3[0];
  for (int f = 0; f < F; ++f) s += xn[(size_t)i * F + f] * w3[f];
  fit[i] = sigmoidf(s);
}
__global__ void k_topk(const float* fit, int* perm, int npg, int k) {
  __shared__ float sv[256];
  __shared__ float rv[256];
  __shared__ int   ri[256];
  const int b = blockIdx.x, t = threadIdx.x;
  sv[t] = (t < npg) ? fit[(size_t)b * npg + t] : -3.4e38f;
  __syncthreads();
  for (int j = 0; j < k; ++j) {
    rv[t] = sv[t];
    ri[t] = t;
    __syncthreads();
    for (int st = 128; st > 0; st >>= 1) {
      if (t < st && rv[t + st] > rv[t]) { rv[t] = rv[t + st]; ri[t] = ri[t + st]; }
      __syncthreads();
    }
    if (t == 0) {
      perm[(size_t)b * k + j] = b * npg + ri[0];
      sv[ri[0]] = -3.4e38f;
    }
    __syncthreads();
  }
}
__global__ void k_gather_scale(const float* xn, const float* fit, const int* perm,
                               float* hp, int nk, int F) {
  long long i = (long long)blockIdx.x * blockDim.x + threadIdx.x;
  if (i >= (long long)nk * F) return;
  int j = (int)(i / F), f = (int)(i % F);
  int p = perm[j];
  hp[i] = xn[(size_t)p * F + f] * fit[p];
}
__global__ void k_pool_mm(const float* hp, float* g, int nb, int k, int F) {
  int i = blockIdx.x * blockDim.x + threadIdx.x;
  if (i >= nb * F) return;
  int b = i / F, f = i % F;
  float mx = -3.4e38f, sm = 0.0f;
  for (int j = 0; j < k; ++j) {
    float v = hp[((size_t)b * k + j) * F + f];
    mx = fmaxf(mx, v);
    sm += v;
  }
  g[i] = 0.5f * (mx + sm / (float)k);
}
__global__ void k_bn2d(float* x, const float* g, const float* b, int R, int C,
                       int relu) {
  int c = blockIdx.x * blockDim.x + threadIdx.x;
  if (c >= C) return;
  float s = 0.0f, q = 0.0f;
  for (int r = 0; r < R; ++r) { float v = x[(size_t)r * C + c]; s += v; q += v * v; }
  float m = s / R, var = q / R - m * m;
  float inv = rsqrtf(var + 1e-5f);
  for (int r = 0; r < R; ++r) {
    float v = (x[(size_t)r * C + c] - m) * inv * g[c] + b[c];
    if (relu) v = fmaxf(v, 0.0f);
    x[(size_t)r * C + c] = v;
  }
}
__global__ void k_bnrows(float* x, const float* g, const float* b, int R, int C,
                         int relu) {
  __shared__ float r1[256], r2[256];
  const int c = blockIdx.x, t = threadIdx.x;
  float s = 0.0f, q = 0.0f;
  for (int r = t; r < R; r += 256) {
    float v = x[(size_t)r * C + c];
    s += v; q += v * v;
  }
  r1[t] = s; r2[t] = q;
  __syncthreads();
  for (int st = 128; st > 0; st >>= 1) {
    if (t < st) { r1[t] += r1[t + st]; r2[t] += r2[t + st]; }
    __syncthreads();
  }
  float m = r1[0] / R, var = r2[0] / R - m * m;
  float inv = rsqrtf(var + 1e-5f);
  for (int r = t; r < R; r += 256) {
    float v = (x[(size_t)r * C + c] - m) * inv * g[c] + b[c];
    if (relu) v = fmaxf(v, 0.0f);
    x[(size_t)r * C + c] = v;
  }
}
__global__ void k_embed(const int* idx, const float* emb, float* out, int n, int D) {
  long long i = (long long)blockIdx.x * blockDim.x + threadIdx.x;
  if (i >= (long long)n * D) return;
  out[i] = emb[(size_t)idx[i / D] * D + (i % D)];
}
__global__ void k_packW(const float* W, float* Wt, int Cout, int K) {
  int i = blockIdx.x * blockDim.x + threadIdx.x;
  if (i >= Cout * K) return;
  int n = i / K, k = i % K;
  Wt[(size_t)k * Cout + n] = W[(size_t)n * K + k];
}
__global__ void k_se_mean(const float* x, float* s, int Bn, int L, int C) {
  int i = blockIdx.x * blockDim.x + threadIdx.x;
  if (i >= Bn * C) return;
  int b = i / C, c = i % C;
  float sm = 0.0f;
  for (int l = 0; l < L; ++l) sm += x[((size_t)b * L + l) * C + c];
  s[i] = sm / (float)L;
}
__global__ void k_se_mlp(const float* s, const float* W1, const float* b1,
                         const float* W2, const float* b2, float* sc,
                         int Bn, int C, int H) {
  int i = blockIdx.x * blockDim.x + threadIdx.x;
  if (i >= Bn * C) return;
  int b = i / C, c = i % C;
  float out = b2[c];
  for (int o = 0; o < H; ++o) {
    float h = b1[o];
    for (int cc = 0; cc < C; ++cc) h += W1[o * C + cc] * s[(size_t)b * C + cc];
    h = fmaxf(h, 0.0f);
    out += W2[c * H + o] * h;
  }
  sc[i] = sigmoidf(out);
}
__global__ void k_se_scale(float* x, const float* sc, int Bn, int L, int C) {
  long long i = (long long)blockIdx.x * blockDim.x + threadIdx.x;
  if (i >= (long long)Bn * L * C) return;
  int c = (int)(i % C);
  int bl = (int)(i / C);
  int b = bl / L;
  x[i] *= sc[(size_t)b * C + c];
}
__global__ void k_pack_cl(const float* x, float* y, int Bn, int L, int C) {
  long long i = (long long)blockIdx.x * blockDim.x + threadIdx.x;
  if (i >= (long long)Bn * C * L) return;
  int b = (int)(i / (C * L));
  int rem = (int)(i % (C * L));
  int c = rem / L, l = rem % L;
  y[i] = x[((size_t)b * L + l) * C + c];
}
__global__ void k_fusion(const float* gt, const float* gd, const float* ct,
                         const float* cd, float* xc, int Bn, int D) {
  int b = blockIdx.x * blockDim.x + threadIdx.x;
  if (b >= Bn) return;
  float mt = -3.4e38f, md = -3.4e38f;
  for (int f = 0; f < D; ++f) {
    mt = fmaxf(mt, gt[b * D + f] * ct[b * D + f]);
    md = fmaxf(md, gd[b * D + f] * cd[b * D + f]);
  }
  float st = 0.0f, sd = 0.0f;
  for (int f = 0; f < D; ++f) {
    st += expf(gt[b * D + f] * ct[b * D + f] - mt);
    sd += expf(gd[b * D + f] * cd[b * D + f] - md);
  }
  float* o = xc + (size_t)b * 8 * D;
  for (int f = 0; f < D; ++f) {
    float at = expf(gt[b * D + f] * ct[b * D + f] - mt) / st;
    float ad = expf(gd[b * D + f] * cd[b * D + f] - md) / sd;
    o[0 * D + f] = gt[b * D + f];
    o[1 * D + f] = gd[b * D + f];
    o[2 * D + f] = ct[b * D + f];
    o[3 * D + f] = cd[b * D + f];
    o[4 * D + f] = gt[b * D + f] * at;
    o[5 * D + f] = gd[b * D + f] * ad;
    o[6 * D + f] = ct[b * D + f] * at;
    o[7 * D + f] = cd[b * D + f] * ad;
  }
}
__global__ void k_out(const float* x, const float* W, const float* b,
                      float* out, int Bn, int K) {
  int i = blockIdx.x * blockDim.x + threadIdx.x;
  if (i >= Bn) return;
  float s = b[0];
  for (int k = 0; k < K; ++k) s += x[(size_t)i * K + k] * W[k];
  out[i] = s;
}

// ---------------------------------------------------------------------------
// Host helpers
// ---------------------------------------------------------------------------
static inline float* wsalloc(char*& p, size_t nfloats) {
  float* r = (float*)p;
  p += ((nfloats * sizeof(float) + 255) / 256) * 256;
  return r;
}

static void gemm(hipStream_t st, const float* A, const float* W, const float* bias,
                 float* C, int M, int N, int K, int act) {
  dim3 g(cdiv(N, 64), cdiv(M, 64));
  k_gemm<0><<<g, 256, 0, st>>>(A, W, bias, C, M, N, K, act, 0, 0);
}

static void run_gcn(hipStream_t st, const float* hin, const float* W,
                    const float* b, float* xw, float* gout, int n, int Fin,
                    int Fout, const int* row, const int* col, int E,
                    const float* dis) {
  gemm(st, hin, W, nullptr, xw, n, Fout, Fin, 0);
  long long tot = (long long)n * Fout;
  k_fill<<<cdiv(tot, 256), 256, 0, st>>>(gout, tot, 0.0f);
  long long te = (long long)E * Fout;
  k_gcn_edges<<<cdiv(te, 256), 256, 0, st>>>(xw, row, col, dis, gout, E, Fout);
  k_gcn_self<<<cdiv(tot, 256), 256, 0, st>>>(xw, dis, b, gout, n, Fout);
}

static void run_pairnorm_relu(hipStream_t st, const float* gout, float* hout,
                              float* cmean, int n, int F) {
  k_colmean<<<F, 256, 0, st>>>(gout, cmean, n, F);
  k_pairnorm_relu<<<cdiv(n, 256), 256, 0, st>>>(gout, hout, cmean, n, F, 1);
}

struct AsapP {
  const float *gnnW, *gnnb, *linW, *linb, *attW, *attb;
  const float *le1W, *le1b, *le2W, *le3W, *le3b;
};

static void run_asap(hipStream_t st, const float* h, const int* row,
                     const int* col, const float* dis, int n, int E, int F,
                     int nb, int npg, int k, float* xw, float* xpool, float* xq,
                     float* xq2, float* s, float* smax, float* dsum, float* xn,
                     float* a, float* b2, float* agg, float* fit, int* perm,
                     float* hp, float* g0, const AsapP& P) {
  const unsigned NEG = 0x007fffffu;  // order-preserving encoding of -inf

  run_gcn(st, h, P.gnnW, P.gnnb, xw, xpool, n, F, F, row, col, E, dis);

  long long nf = (long long)n * F;
  long long te = (long long)(E + n) * F;
  int Et = E + n;

  k_init_u<<<cdiv(nf, 256), 256, 0, st>>>((unsigned*)xq, nf, NEG);
  k_segmax_feat<<<cdiv(te, 256), 256, 0, st>>>(xpool, row, col, (unsigned*)xq, E, n, F);
  k_decode_u<<<cdiv(nf, 256), 256, 0, st>>>((unsigned*)xq, nf);

  gemm(st, xq, P.linW, P.linb, xq2, n, F, F, 0);

  k_asap_att<<<cdiv(Et, 256), 256, 0, st>>>(xq2, xpool, row, col, P.attW, P.attb, s, E, n, F);

  k_init_u<<<cdiv((long long)n, 256), 256, 0, st>>>((unsigned*)smax, n, NEG);
  k_segmax_scalar<<<cdiv(Et, 256), 256, 0, st>>>(s, col, (unsigned*)smax, E, n);
  k_decode_u<<<cdiv((long long)n, 256), 256, 0, st>>>((unsigned*)smax, n);
  k_fill<<<cdiv((long long)n, 256), 256, 0, st>>>(dsum, n, 0.0f);
  k_segexp<<<cdiv(Et, 256), 256, 0, st>>>(s, smax, dsum, col, E, n);
  k_segdiv<<<cdiv(Et, 256), 256, 0, st>>>(s, dsum, col, E, n);

  k_fill<<<cdiv(nf, 256), 256, 0, st>>>(xn, nf, 0.0f);
  k_weighted_scatter<<<cdiv(te, 256), 256, 0, st>>>(h, s, row, col, xn, E, n, F);

  k_le_ab<<<cdiv((long long)n, 256), 256, 0, st>>>(xn, P.le1W, P.le1b, P.le2W, a, b2, n, F);
  k_fill<<<cdiv((long long)n, 256), 256, 0, st>>>(agg, n, 0.0f);
  k_le_agg<<<cdiv(Et, 256), 256, 0, st>>>(a, b2, row, col, agg, E, n);
  k_fit<<<cdiv((long long)n, 256), 256, 0, st>>>(xn, agg, P.le3W, P.le3b, fit, n, F);

  k_topk<<<nb, 256, 0, st>>>(fit, perm, npg, k);
  int nk = nb * k;
  k_gather_scale<<<cdiv((long long)nk * F, 256), 256, 0, st>>>(xn, fit, perm, hp, nk, F);
  k_pool_mm<<<cdiv((long long)nb * F, 256), 256, 0, st>>>(hp, g0, nb, k, F);
}

static void run_conv(hipStream_t st, const float* X, int mode, int Cin, int L,
                     int Cout, const float* Wc, const float* bc, float* Wt,
                     float* Y, int Bn) {
  int K = Cin * 3, M = Bn * L;
  k_packW<<<cdiv((long long)Cout * K, 256), 256, 0, st>>>(Wc, Wt, Cout, K);
  dim3 g(cdiv(Cout, 64), cdiv(M, 64));
  if (mode == 1)
    k_gemm<1><<<g, 256, 0, st>>>(X, Wt, bc, Y, M, Cout, K, 0, Cin, L);
  else
    k_gemm<2><<<g, 256, 0, st>>>(X, Wt, bc, Y, M, Cout, K, 0, Cin, L);
}

// ---------------------------------------------------------------------------
// Orchestration
// ---------------------------------------------------------------------------
extern "C" void kernel_launch(void* const* d_in, const int* in_sizes, int n_in,
                              void* d_out, int out_size, void* d_ws,
                              size_t ws_size, hipStream_t stream) {
  (void)in_sizes; (void)n_in; (void)out_size; (void)ws_size;
  int ii = 0;
  auto NXT = [&]() -> const float* { return (const float*)d_in[ii++]; };

  const float* x  = NXT();
  const float* tx = NXT();
  // params (setup_inputs insertion order)
  const float *gd1_W = NXT(), *gd1_b = NXT(), *gd2_W = NXT(), *gd2_b = NXT();
  const float *gd3_W = NXT(), *gd3_b = NXT();
  const float *apd_gnn_W = NXT(), *apd_gnn_b = NXT();
  const float *apd_lin_W = NXT(), *apd_lin_b = NXT();
  const float *apd_att_W = NXT(), *apd_att_b = NXT();
  const float *apd_le1_W = NXT(), *apd_le1_b = NXT();
  const float *apd_le2_W = NXT();
  const float *apd_le3_W = NXT(), *apd_le3_b = NXT();
  const float *fcdg_W = NXT(), *fcdg_b = NXT();
  const float *bn_fcdg_g = NXT(), *bn_fcdg_b = NXT();
  const float *emb_xd = NXT();
  const float *cs1_W = NXT(), *cs1_b = NXT(), *bn_cs1_g = NXT(), *bn_cs1_b = NXT();
  const float *cs2_W = NXT(), *cs2_b = NXT(), *bn_cs2_g = NXT(), *bn_cs2_b = NXT();
  const float *dse1_W = NXT(), *dse1_b = NXT(), *dse2_W = NXT(), *dse2_b = NXT();
  const float *fcs1_W = NXT(), *fcs1_b = NXT(), *bn_fcs1_g = NXT(), *bn_fcs1_b = NXT();
  const float *fcs2_W = NXT(), *fcs2_b = NXT(), *bn_fcs2_g = NXT(), *bn_fcs2_b = NXT();
  const float *gt1_W = NXT(), *gt1_b = NXT(), *gt2_W = NXT(), *gt2_b = NXT();
  const float *gt3_W = NXT(), *gt3_b = NXT();
  const float *apt_gnn_W = NXT(), *apt_gnn_b = NXT();
  const float *apt_lin_W = NXT(), *apt_lin_b = NXT();
  const float *apt_att_W = NXT(), *apt_att_b = NXT();
  const float *apt_le1_W = NXT(), *apt_le1_b = NXT();
  const float *apt_le2_W = NXT();
  const float *apt_le3_W = NXT(), *apt_le3_b = NXT();
  const float *fctg_W = NXT(), *fctg_b = NXT();
  const float *bn_fctg_g = NXT(), *bn_fctg_b = NXT();
  const float *emb_xt = NXT();
  const float *cp1_W = NXT(), *cp1_b = NXT(), *bn_cp1_g = NXT(), *bn_cp1_b = NXT();
  const float *cp2_W = NXT(), *cp2_b = NXT(), *bn_cp2_g = NXT(), *bn_cp2_b = NXT();
  const float *cp3_W = NXT(), *cp3_b = NXT(), *bn_cp3_g = NXT(), *bn_cp3_b = NXT();
  const float *tse1_W = NXT(), *tse1_b = NXT(), *tse2_W = NXT(), *tse2_b = NXT();
  const float *fcp1_W = NXT(), *fcp1_b = NXT(), *bn_fcp1_g = NXT(), *bn_fcp1_b = NXT();
  const float *fcp2_W = NXT(), *fcp2_b = NXT(), *bn_fcp2_g = NXT(), *bn_fcp2_b = NXT();
  const float *fcc1_W = NXT(), *fcc1_b = NXT(), *bn_fcc1_g = NXT(), *bn_fcc1_b = NXT();
  const float *fcc2_W = NXT(), *fcc2_b = NXT(), *bn_fcc2_g = NXT(), *bn_fcc2_b = NXT();
  const float *out_W = NXT(), *out_b = NXT();
  const int* edge_index   = (const int*)d_in[ii++];
  const int* smiles       = (const int*)d_in[ii++];
  const int* t_edge_index = (const int*)d_in[ii++];
  const int* target       = (const int*)d_in[ii++];

  const int Bn = 256, EMB = 128;
  hipStream_t st = stream;

  char* wp = (char*)d_ws;
  // persistent across branches
  float* gd_  = wsalloc(wp, (size_t)Bn * 128);
  float* cd_  = wsalloc(wp, (size_t)Bn * 128);
  float* gt_  = wsalloc(wp, (size_t)Bn * 128);
  float* ct_  = wsalloc(wp, (size_t)Bn * 128);
  float* xc   = wsalloc(wp, (size_t)Bn * 1024);
  float* f1   = wsalloc(wp, (size_t)Bn * 1024);
  float* f2   = wsalloc(wp, (size_t)Bn * 512);
  float* cmean = wsalloc(wp, 256);
  char* mark = wp;

  // ===================== drug graph branch =====================
  {
    const int n = 16384, E = 65536, Fg = 78, F3 = 234, npg = 64, kk = 32;
    const int* row = edge_index;
    const int* col = edge_index + E;
    float* deg = wsalloc(wp, n);
    float* dis = wsalloc(wp, n);
    k_fill<<<cdiv((long long)n, 256), 256, 0, st>>>(deg, n, 1.0f);
    k_degree<<<cdiv((long long)E, 256), 256, 0, st>>>(col, E, deg);
    k_invsqrt<<<cdiv((long long)n, 256), 256, 0, st>>>(deg, dis, n);

    float* xw   = wsalloc(wp, (size_t)n * F3);
    float* gout = wsalloc(wp, (size_t)n * F3);
    float* h1   = wsalloc(wp, (size_t)n * Fg);
    float* h2   = wsalloc(wp, (size_t)n * Fg);
    float* h3   = wsalloc(wp, (size_t)n * Fg);
    run_gcn(st, x,  gd1_W, gd1_b, xw, gout, n, Fg, Fg, row, col, E, dis);
    run_pairnorm_relu(st, gout, h1, cmean, n, Fg);
    run_gcn(st, h1, gd2_W, gd2_b, xw, gout, n, Fg, Fg, row, col, E, dis);
    run_pairnorm_relu(st, gout, h2, cmean, n, Fg);
    run_gcn(st, h2, gd3_W, gd3_b, xw, gout, n, Fg, Fg, row, col, E, dis);
    run_pairnorm_relu(st, gout, h3, cmean, n, Fg);

    float* hcat = wsalloc(wp, (size_t)n * F3);
    k_concat3<<<cdiv((long long)n * F3, 256), 256, 0, st>>>(h1, h2, h3, hcat, n, Fg);

    float* xpool = gout;  // reuse (sized n*F3)
    float* xq   = wsalloc(wp, (size_t)n * F3);
    float* xq2  = wsalloc(wp, (size_t)n * F3);
    float* s    = wsalloc(wp, (size_t)(E + n));
    float* smax = wsalloc(wp, n);
    float* dsum = wsalloc(wp, n);
    float* xn   = wsalloc(wp, (size_t)n * F3);
    float* av   = wsalloc(wp, n);
    float* b2v  = wsalloc(wp, n);
    float* agg  = wsalloc(wp, n);
    float* fit  = wsalloc(wp, n);
    int*   perm = (int*)wsalloc(wp, (size_t)Bn * kk);
    float* hp   = wsalloc(wp, (size_t)Bn * kk * F3);
    float* g0   = wsalloc(wp, (size_t)Bn * F3);

    AsapP P = {apd_gnn_W, apd_gnn_b, apd_lin_W, apd_lin_b, apd_att_W, apd_att_b,
               apd_le1_W, apd_le1_b, apd_le2_W, apd_le3_W, apd_le3_b};
    run_asap(st, hcat, row, col, dis, n, E, F3, Bn, npg, kk, xw, xpool, xq, xq2,
             s, smax, dsum, xn, av, b2v, agg, fit, perm, hp, g0, P);

    gemm(st, g0, fcdg_W, fcdg_b, gd_, Bn, 128, F3, 0);
    k_bn2d<<<1, 128, 0, st>>>(gd_, bn_fcdg_g, bn_fcdg_b, Bn, 128, 1);
  }
  wp = mark;

  // ===================== SMILES CNN branch =====================
  {
    const int L = 128;
    float* e  = wsalloc(wp, (size_t)Bn * 100 * EMB);
    k_embed<<<cdiv((long long)Bn * 100 * EMB, 256), 256, 0, st>>>(smiles, emb_xd, e, Bn * 100, EMB);
    float* Wt = wsalloc(wp, 300 * 64);
    float* c1 = wsalloc(wp, (size_t)Bn * L * 64);
    run_conv(st, e, 1, 100, L, 64, cs1_W, cs1_b, Wt, c1, Bn);
    k_bnrows<<<64, 256, 0, st>>>(c1, bn_cs1_g, bn_cs1_b, Bn * L, 64, 1);
    float* c2 = wsalloc(wp, (size_t)Bn * L * 32);
    run_conv(st, c1, 2, 64, L, 32, cs2_W, cs2_b, Wt, c2, Bn);
    k_bnrows<<<32, 256, 0, st>>>(c2, bn_cs2_g, bn_cs2_b, Bn * L, 32, 1);
    // SE
    float* sm = wsalloc(wp, (size_t)Bn * 32);
    float* sc = wsalloc(wp, (size_t)Bn * 32);
    k_se_mean<<<cdiv((long long)Bn * 32, 256), 256, 0, st>>>(c2, sm, Bn, L, 32);
    k_se_mlp<<<cdiv((long long)Bn * 32, 256), 256, 0, st>>>(sm, dse1_W, dse1_b, dse2_W, dse2_b, sc, Bn, 32, 2);
    k_se_scale<<<cdiv((long long)Bn * L * 32, 256), 256, 0, st>>>(c2, sc, Bn, L, 32);
    float* flat = wsalloc(wp, (size_t)Bn * 4096);
    k_pack_cl<<<cdiv((long long)Bn * 4096, 256), 256, 0, st>>>(c2, flat, Bn, L, 32);
    float* fc1 = wsalloc(wp, (size_t)Bn * 4096);
    gemm(st, flat, fcs1_W, fcs1_b, fc1, Bn, 4096, 4096, 1);
    k_bn2d<<<16, 256, 0, st>>>(fc1, bn_fcs1_g, bn_fcs1_b, Bn, 4096, 0);
    gemm(st, fc1, fcs2_W, fcs2_b, cd_, Bn, 128, 4096, 1);
    k_bn2d<<<1, 128, 0, st>>>(cd_, bn_fcs2_g, bn_fcs2_b, Bn, 128, 0);
  }
  wp = mark;

  // ===================== target graph branch =====================
  {
    const int n = 65536, E = 262144, Fg = 54, F3 = 162, npg = 256, kk = 128;
    const int* row = t_edge_index;
    const int* col = t_edge_index + E;
    float* deg = wsalloc(wp, n);
    float* dis = wsalloc(wp, n);
    k_fill<<<cdiv((long long)n, 256), 256, 0, st>>>(deg, n, 1.0f);
    k_degree<<<cdiv((long long)E, 256), 256, 0, st>>>(col, E, deg);
    k_invsqrt<<<cdiv((long long)n, 256), 256, 0, st>>>(deg, dis, n);

    float* xw   = wsalloc(wp, (size_t)n * F3);
    float* gout = wsalloc(wp, (size_t)n * F3);
    float* h1   = wsalloc(wp, (size_t)n * Fg);
    float* h2   = wsalloc(wp, (size_t)n * Fg);
    float* h3   = wsalloc(wp, (size_t)n * Fg);
    run_gcn(st, tx, gt1_W, gt1_b, xw, gout, n, Fg, Fg, row, col, E, dis);
    run_pairnorm_relu(st, gout, h1, cmean, n, Fg);
    run_gcn(st, h1, gt2_W, gt2_b, xw, gout, n, Fg, Fg, row, col, E, dis);
    run_pairnorm_relu(st, gout, h2, cmean, n, Fg);
    run_gcn(st, h2, gt3_W, gt3_b, xw, gout, n, Fg, Fg, row, col, E, dis);
    run_pairnorm_relu(st, gout, h3, cmean, n, Fg);

    float* hcat = wsalloc(wp, (size_t)n * F3);
    k_concat3<<<cdiv((long long)n * F3, 256), 256, 0, st>>>(h1, h2, h3, hcat, n, Fg);

    float* xpool = gout;
    float* xq   = wsalloc(wp, (size_t)n * F3);
    float* xq2  = wsalloc(wp, (size_t)n * F3);
    float* s    = wsalloc(wp, (size_t)(E + n));
    float* smax = wsalloc(wp, n);
    float* dsum = wsalloc(wp, n);
    float* xn   = wsalloc(wp, (size_t)n * F3);
    float* av   = wsalloc(wp, n);
    float* b2v  = wsalloc(wp, n);
    float* agg  = wsalloc(wp, n);
    float* fit  = wsalloc(wp, n);
    int*   perm = (int*)wsalloc(wp, (size_t)Bn * kk);
    float* hp   = wsalloc(wp, (size_t)Bn * kk * F3);
    float* g0   = wsalloc(wp, (size_t)Bn * F3);

    AsapP P = {apt_gnn_W, apt_gnn_b, apt_lin_W, apt_lin_b, apt_att_W, apt_att_b,
               apt_le1_W, apt_le1_b, apt_le2_W, apt_le3_W, apt_le3_b};
    run_asap(st, hcat, row, col, dis, n, E, F3, Bn, npg, kk, xw, xpool, xq, xq2,
             s, smax, dsum, xn, av, b2v, agg, fit, perm, hp, g0, P);

    gemm(st, g0, fctg_W, fctg_b, gt_, Bn, 128, F3, 0);
    k_bn2d<<<1, 128, 0, st>>>(gt_, bn_fctg_g, bn_fctg_b, Bn, 128, 1);
  }
  wp = mark;

  // ===================== protein CNN branch =====================
  {
    const int L = 128;
    float* e  = wsalloc(wp, (size_t)Bn * 1000 * EMB);
    k_embed<<<cdiv((long long)Bn * 1000 * EMB, 256), 256, 0, st>>>(target, emb_xt, e, Bn * 1000, EMB);
    float* Wt = wsalloc(wp, 3000 * 128);
    float* c1 = wsalloc(wp, (size_t)Bn * L * 128);
    run_conv(st, e, 1, 1000, L, 128, cp1_W, cp1_b, Wt, c1, Bn);
    k_bnrows<<<128, 256, 0, st>>>(c1, bn_cp1_g, bn_cp1_b, Bn * L, 128, 1);
    float* c2 = wsalloc(wp, (size_t)Bn * L * 64);
    run_conv(st, c1, 2, 128, L, 64, cp2_W, cp2_b, Wt, c2, Bn);
    k_bnrows<<<64, 256, 0, st>>>(c2, bn_cp2_g, bn_cp2_b, Bn * L, 64, 1);
    float* c3 = wsalloc(wp, (size_t)Bn * L * 32);
    run_conv(st, c2, 2, 64, L, 32, cp3_W, cp3_b, Wt, c3, Bn);
    k_bnrows<<<32, 256, 0, st>>>(c3, bn_cp3_g, bn_cp3_b, Bn * L, 32, 1);
    // SE
    float* sm = wsalloc(wp, (size_t)Bn * 32);
    float* sc = wsalloc(wp, (size_t)Bn * 32);
    k_se_mean<<<cdiv((long long)Bn * 32, 256), 256, 0, st>>>(c3, sm, Bn, L, 32);
    k_se_mlp<<<cdiv((long long)Bn * 32, 256), 256, 0, st>>>(sm, tse1_W, tse1_b, tse2_W, tse2_b, sc, Bn, 32, 2);
    k_se_scale<<<cdiv((long long)Bn * L * 32, 256), 256, 0, st>>>(c3, sc, Bn, L, 32);
    float* flat = wsalloc(wp, (size_t)Bn * 4096);
    k_pack_cl<<<cdiv((long long)Bn * 4096, 256), 256, 0, st>>>(c3, flat, Bn, L, 32);
    float* fc1 = wsalloc(wp, (size_t)Bn * 4096);
    gemm(st, flat, fcp1_W, fcp1_b, fc1, Bn, 4096, 4096, 1);
    k_bn2d<<<16, 256, 0, st>>>(fc1, bn_fcp1_g, bn_fcp1_b, Bn, 4096, 0);
    gemm(st, fc1, fcp2_W, fcp2_b, ct_, Bn, 128, 4096, 1);
    k_bn2d<<<1, 128, 0, st>>>(ct_, bn_fcp2_g, bn_fcp2_b, Bn, 128, 0);
  }
  wp = mark;

  // ===================== fusion head =====================
  k_fusion<<<1, 256, 0, st>>>(gt_, gd_, ct_, cd_, xc, Bn, 128);
  gemm(st, xc, fcc1_W, fcc1_b, f1, Bn, 1024, 1024, 1);
  k_bn2d<<<4, 256, 0, st>>>(f1, bn_fcc1_g, bn_fcc1_b, Bn, 1024, 0);
  gemm(st, f1, fcc2_W, fcc2_b, f2, Bn, 512, 1024, 1);
  k_bn2d<<<2, 256, 0, st>>>(f2, bn_fcc2_g, bn_fcc2_b, Bn, 512, 0);
  k_out<<<1, 256, 0, st>>>(f2, out_W, out_b, (float*)d_out, Bn, 512);
}
